// NodeSchNetWrapper_28243704938498
// MI455X (gfx1250) — compile-verified, matrix operand order checked by reading, not run on hardware
//
#include <hip/hip_runtime.h>
#include <hip/hip_bf16.h>

// ---------------- constants ----------------
#define HID   600
#define HIDP  640          // 20*32 = 10*64 : K regions of 64
#define NG    50
#define NGP   64
#define NI    6
#define NNODE 8000
#define NEDGE 128000
#define NBATCH 128
#define ECH   32000        // edge chunk
#define NCH   4
#define LOG2F_ 0.69314718055994530942f
#define CUTF  10.0f

typedef __attribute__((ext_vector_type(16))) __bf16 v16bf;
typedef __attribute__((ext_vector_type(8)))  float  v8f;

__device__ __forceinline__ float sspf(float x) {
    // softplus(x) - log(2), numerically stable
    float sp = fmaxf(x, 0.f) + __logf(1.f + __expf(-fabsf(x)));
    return sp - LOG2F_;
}

__device__ __forceinline__ void atomicAddF(float* p, float v) {
    __hip_atomic_fetch_add(p, v, __ATOMIC_RELAXED, __HIP_MEMORY_SCOPE_AGENT);
}

// ---------------- WMMA GEMM core ----------------
// Block: 256 threads = 8 waves, wave grid 4(M) x 2(N) of 16x16 tiles -> 64x32 block tile.
// K staged in regions of 64 -> 2 chained v_wmma per barrier region.
// A: [M, K] bf16 row-major (lda), B: [K, N] bf16 row-major (ldb).
// REQUIREMENTS (all launches satisfy them): M % 64 == 0, K % 64 == 0,
// lda/ldb multiples of 32 -> every 8-elem segment is 16B aligned (pure b128 staging).
// Fragment layout per ISA 7.12.2 (16-bit A 16x32): lane = m%16 (n%16 for B),
// lane-half selects K {0..7,16..23} vs {8..15,24..31}; each VGPR packs K,K+1.
__device__ __forceinline__ v8f gemm_core(const __bf16* __restrict__ A,
                                         const __bf16* __restrict__ B,
                                         int K, int lda, int ldb,
                                         int mBase, int nBase,
                                         __bf16* As /*64*64*/, __bf16* Bst /*32*64 n-major*/) {
    const int tid  = threadIdx.x;
    const int lane = tid & 31;
    const int wid  = tid >> 5;
    const int wm   = wid & 3;       // wave tile row (0..3)
    const int wn   = wid >> 2;      // wave tile col (0..1)
    const int hi   = lane >> 4;     // lane half
    const int l16  = lane & 15;

    // A staging: 64 rows x 64 K = 512 x 16B chunks; thread does rows r and r+32
    const int arow = tid >> 3;            // 0..31
    const int acol = (tid & 7) << 3;      // {0,8,...,56}
    // B staging: 64 k-rows x 32 n = 256 x 16B chunks; one per thread
    const int bk = tid >> 2;              // 0..63
    const int bc = (tid & 3) << 3;        // {0,8,16,24}

    const __bf16* __restrict__ Ap0 = A + (size_t)(mBase + arow) * lda + acol;
    const __bf16* __restrict__ Ap1 = Ap0 + (size_t)32 * lda;
    const __bf16* __restrict__ Bp  = B + (size_t)bk * ldb + nBase + bc;

    v8f acc = {};
    for (int k0 = 0; k0 < K; k0 += 64) {
        __syncthreads();
        *(uint4*)(As + arow * 64 + acol)        = *(const uint4*)(Ap0 + k0);
        *(uint4*)(As + (arow + 32) * 64 + acol) = *(const uint4*)(Ap1 + k0);
        {   // B: load 8 consecutive n for k-row bk, transpose into n-major LDS
            union { uint4 q; __bf16 h[8]; } ub;
            ub.q = *(const uint4*)(Bp + (size_t)k0 * ldb);
#pragma unroll
            for (int j = 0; j < 8; ++j) Bst[(bc + j) * 64 + bk] = ub.h[j];
        }
        if (k0 + 64 < K) {  // prefetch next region (global_prefetch_b8)
            __builtin_prefetch(Ap0 + k0 + 64, 0, 1);
            __builtin_prefetch(Ap1 + k0 + 64, 0, 1);
            __builtin_prefetch(Bp + (size_t)(k0 + 64) * ldb, 0, 1);
        }
        __syncthreads();

        const unsigned* ar2 = (const unsigned*)(As  + (wm * 16 + l16) * 64);
        const unsigned* br2 = (const unsigned*)(Bst + (wn * 16 + l16) * 64);
#pragma unroll
        for (int s = 0; s < 2; ++s) {       // two K=32 sub-steps per region
            union { v16bf v; unsigned u[8]; } fa, fb;
#pragma unroll
            for (int v = 0; v < 8; ++v) {
                // dword p covers K = 2p; pair base = (v&3)*2 + (v>=4)*16 + hi*8 (+32*s)
                int p = (v & 3) + ((v >> 2) << 3) + (hi << 2) + (s << 4);
                fa.u[v] = ar2[p];
                fb.u[v] = br2[p];
            }
            acc = __builtin_amdgcn_wmma_f32_16x16x32_bf16(
                      false, fa.v, false, fb.v, (short)0, acc, false, false);
        }
    }
    return acc;
}

// EPI: 0 = outF = acc + bias
//      1 = outB = bf16( ssp(acc + bias) )
//      2 = v = acc + bias + res;  outF = v; outB = bf16(v)   (residual update)
template <int EPI>
__global__ void k_gemm(const __bf16* __restrict__ A, const __bf16* __restrict__ B,
                       const float* __restrict__ bias, const float* __restrict__ res,
                       float* __restrict__ outF, __bf16* __restrict__ outB,
                       int M, int K, int lda, int ldb, int ldo, int NW) {
    __shared__ __bf16 As[64 * 64];
    __shared__ __bf16 Bst[32 * 64];
    const int mBase = blockIdx.x * 64;
    const int nBase = blockIdx.y * 32;
    v8f acc = gemm_core(A, B, K, lda, ldb, mBase, nBase, As, Bst);

    const int tid = threadIdx.x, lane = tid & 31, wid = tid >> 5;
    const int wm = wid & 3, wn = wid >> 2, hi = lane >> 4, l16 = lane & 15;
    const int gn = nBase + wn * 16 + l16;
#pragma unroll
    for (int j = 0; j < 8; ++j) {
        int gm = mBase + wm * 16 + j + hi * 8;   // C/D layout: vgpr j -> M=j (+8 upper half)
        if (gm < M && gn < NW) {
            float v = acc[j];
            if (bias && gn < HID) v += bias[gn];
            if (EPI == 1) v = sspf(v);
            if (EPI == 2) v += res[(size_t)gm * ldo + gn];
            if (EPI == 0 || EPI == 2) outF[(size_t)gm * ldo + gn] = v;
            if (EPI == 1 || EPI == 2) outB[(size_t)gm * ldo + gn] = (__bf16)v;
        }
    }
}

// Fused: W = t1 @ Wm2 + bm2 ; msg = x1[src] * W * C ; agg[dst] += msg  (scatter epilogue)
__global__ void k_gemm_scatter(const __bf16* __restrict__ A, const __bf16* __restrict__ B,
                               const float* __restrict__ bias,
                               const float* __restrict__ Cg, const float* __restrict__ x1,
                               const int* __restrict__ src, const int* __restrict__ dst,
                               float* __restrict__ agg, int Mchunk, int ebase) {
    __shared__ __bf16 As[64 * 64];
    __shared__ __bf16 Bst[32 * 64];
    const int mBase = blockIdx.x * 64;
    const int nBase = blockIdx.y * 32;
    v8f acc = gemm_core(A, B, HIDP, HIDP, HIDP, mBase, nBase, As, Bst);

    const int tid = threadIdx.x, lane = tid & 31, wid = tid >> 5;
    const int wm = wid & 3, wn = wid >> 2, hi = lane >> 4, l16 = lane & 15;
    const int gn = nBase + wn * 16 + l16;
#pragma unroll
    for (int j = 0; j < 8; ++j) {
        int gm = mBase + wm * 16 + j + hi * 8;
        if (gm < Mchunk && gn < HID) {
            float wv = acc[j] + bias[gn];
            int ge = ebase + gm;
            float val = wv * Cg[ge] * x1[(size_t)src[ge] * HIDP + gn];
            atomicAddF(&agg[(size_t)dst[ge] * HIDP + gn], val);
        }
    }
}

// ---------------- elementwise / prep kernels ----------------
__global__ void k_edge_geom(const float* __restrict__ pos, const int* __restrict__ src,
                            const int* __restrict__ dst, float* __restrict__ d,
                            float* __restrict__ Cg) {
    int e = blockIdx.x * blockDim.x + threadIdx.x;
    if (e >= NEDGE) return;
    int s = src[e], t = dst[e];
    float dx = pos[s * 3 + 0] - pos[t * 3 + 0];
    float dy = pos[s * 3 + 1] - pos[t * 3 + 1];
    float dz = pos[s * 3 + 2] - pos[t * 3 + 2];
    float dd = sqrtf(dx * dx + dy * dy + dz * dz);
    d[e] = dd;
    float c = 0.5f * (__cosf(dd * 3.14159265358979323846f / CUTF) + 1.0f);
    Cg[e] = (dd < CUTF) ? c : 0.0f;
}

__global__ void k_smear(const float* __restrict__ d, __bf16* __restrict__ ea) {
    long idx = (long)blockIdx.x * blockDim.x + threadIdx.x;
    if (idx >= (long)NEDGE * NGP) return;
    int e = (int)(idx >> 6), g = (int)(idx & 63);
    float step  = CUTF / (float)(NG - 1);
    float coeff = -0.5f / (step * step);
    float v = 0.0f;
    if (g < NG) {
        float t = d[e] - (float)g * step;
        v = __expf(coeff * t * t);
    }
    ea[idx] = (__bf16)v;
}

__global__ void k_pad_w(const float* __restrict__ in, __bf16* __restrict__ out,
                        int Rin, int Cin, int Rout, int Cout) {
    long idx = (long)blockIdx.x * blockDim.x + threadIdx.x;
    if (idx >= (long)Rout * Cout) return;
    int r = (int)(idx / Cout), c = (int)(idx % Cout);
    float v = (r < Rin && c < Cin) ? in[(size_t)r * Cin + c] : 0.0f;
    out[idx] = (__bf16)v;
}

__global__ void k_gather_h(const int* __restrict__ z, const float* __restrict__ emb,
                           float* __restrict__ hf, __bf16* __restrict__ hb) {
    long idx = (long)blockIdx.x * blockDim.x + threadIdx.x;
    if (idx >= (long)NNODE * HIDP) return;
    int n = (int)(idx / HIDP), j = (int)(idx % HIDP);
    float v = (j < HID) ? emb[(size_t)z[n] * HID + j] : 0.0f;
    hf[idx] = v;
    hb[idx] = (__bf16)v;
}

__global__ void k_zero(float* __restrict__ p, long n) {
    long idx = (long)blockIdx.x * blockDim.x + threadIdx.x;
    if (idx < n) p[idx] = 0.0f;
}

__global__ void k_f2b(const float* __restrict__ in, __bf16* __restrict__ out, long n) {
    long idx = (long)blockIdx.x * blockDim.x + threadIdx.x;
    if (idx < n) out[idx] = (__bf16)in[idx];
}

__global__ void k_pool(const float* __restrict__ hf, const int* __restrict__ batch,
                       float* __restrict__ psum, float* __restrict__ cnt) {
    long idx = (long)blockIdx.x * blockDim.x + threadIdx.x;
    if (idx >= (long)NNODE * HIDP) return;
    int n = (int)(idx / HIDP), j = (int)(idx % HIDP);
    int b = batch[n];
    atomicAddF(&psum[(size_t)b * HIDP + j], hf[idx]);
    if (j == 0) atomicAddF(&cnt[b], 1.0f);
}

__global__ void k_pdiv(const float* __restrict__ psum, const float* __restrict__ cnt,
                       __bf16* __restrict__ pbf) {
    long idx = (long)blockIdx.x * blockDim.x + threadIdx.x;
    if (idx >= (long)NBATCH * HIDP) return;
    int b = (int)(idx / HIDP);
    pbf[idx] = (__bf16)(psum[idx] / fmaxf(cnt[b], 1.0f));
}

// ---------------- host launcher ----------------
static inline int cdiv_i(long a, long b) { return (int)((a + b - 1) / b); }
static inline size_t alg(size_t x) { return (x + 255) & ~(size_t)255; }

extern "C" void kernel_launch(void* const* d_in, const int* in_sizes, int n_in,
                              void* d_out, int out_size, void* d_ws, size_t ws_size,
                              hipStream_t stream) {
    const int*   z     = (const int*)d_in[0];
    const float* pos   = (const float*)d_in[1];
    const int*   batch = (const int*)d_in[2];
    const int*   eidx  = (const int*)d_in[3];
    const float* emb   = (const float*)d_in[4];
    const float* Wm1   = (const float*)d_in[5];
    const float* bm1   = (const float*)d_in[6];
    const float* Wm2   = (const float*)d_in[7];
    const float* bm2   = (const float*)d_in[8];
    const float* Wl1   = (const float*)d_in[9];
    const float* Wl2   = (const float*)d_in[10];
    const float* bl2   = (const float*)d_in[11];
    const float* Wlin  = (const float*)d_in[12];
    const float* blin  = (const float*)d_in[13];
    const float* Wpool = (const float*)d_in[14];
    const float* bpool = (const float*)d_in[15];
    const int* srcp = eidx;
    const int* dstp = eidx + NEDGE;
    float* outp = (float*)d_out;

    // workspace carve
    char* w = (char*)d_ws;
    auto take = [&](size_t bytes) { char* p = w; w += alg(bytes); return p; };
    float*  dbuf  = (float*)take((size_t)NEDGE * 4);
    float*  Cg    = (float*)take((size_t)NEDGE * 4);
    __bf16* eab   = (__bf16*)take((size_t)NEDGE * NGP * 2);
    __bf16* t1b   = (__bf16*)take((size_t)ECH * HIDP * 2);
    float*  x1    = (float*)take((size_t)NNODE * HIDP * 4);
    float*  agg   = (float*)take((size_t)NNODE * HIDP * 4);
    __bf16* aggb  = (__bf16*)take((size_t)NNODE * HIDP * 2);
    float*  hf    = (float*)take((size_t)NNODE * HIDP * 4);
    __bf16* hb    = (__bf16*)take((size_t)NNODE * HIDP * 2);
    __bf16* tb    = (__bf16*)take((size_t)NNODE * HIDP * 2);
    __bf16* wm1b  = (__bf16*)take((size_t)NI * NGP * HIDP * 2);
    __bf16* wm2b  = (__bf16*)take((size_t)NI * HIDP * HIDP * 2);
    __bf16* wl1b  = (__bf16*)take((size_t)NI * HIDP * HIDP * 2);
    __bf16* wl2b  = (__bf16*)take((size_t)NI * HIDP * HIDP * 2);
    __bf16* wlinb = (__bf16*)take((size_t)NI * HIDP * HIDP * 2);
    __bf16* wpb   = (__bf16*)take((size_t)HIDP * HIDP * 2);
    float*  psum  = (float*)take((size_t)NBATCH * HIDP * 4);
    float*  cnt   = (float*)take((size_t)NBATCH * 4);
    __bf16* pbf   = (__bf16*)take((size_t)NBATCH * HIDP * 2);

    const int T = 256;
    // weight conversion (bf16 + zero padding)
    for (int i = 0; i < NI; ++i) {
        k_pad_w<<<cdiv_i((long)NGP * HIDP, T), T, 0, stream>>>(
            Wm1 + (size_t)i * NG * HID, wm1b + (size_t)i * NGP * HIDP, NG, HID, NGP, HIDP);
        k_pad_w<<<cdiv_i((long)HIDP * HIDP, T), T, 0, stream>>>(
            Wm2 + (size_t)i * HID * HID, wm2b + (size_t)i * HIDP * HIDP, HID, HID, HIDP, HIDP);
        k_pad_w<<<cdiv_i((long)HIDP * HIDP, T), T, 0, stream>>>(
            Wl1 + (size_t)i * HID * HID, wl1b + (size_t)i * HIDP * HIDP, HID, HID, HIDP, HIDP);
        k_pad_w<<<cdiv_i((long)HIDP * HIDP, T), T, 0, stream>>>(
            Wl2 + (size_t)i * HID * HID, wl2b + (size_t)i * HIDP * HIDP, HID, HID, HIDP, HIDP);
        k_pad_w<<<cdiv_i((long)HIDP * HIDP, T), T, 0, stream>>>(
            Wlin + (size_t)i * HID * HID, wlinb + (size_t)i * HIDP * HIDP, HID, HID, HIDP, HIDP);
    }
    k_pad_w<<<cdiv_i((long)HIDP * HIDP, T), T, 0, stream>>>(Wpool, wpb, HID, HID, HIDP, HIDP);

    // edge geometry + gaussian smearing
    k_edge_geom<<<cdiv_i(NEDGE, T), T, 0, stream>>>(pos, srcp, dstp, dbuf, Cg);
    k_smear<<<cdiv_i((long)NEDGE * NGP, T), T, 0, stream>>>(dbuf, eab);

    // h = emb[z]
    k_gather_h<<<cdiv_i((long)NNODE * HIDP, T), T, 0, stream>>>(z, emb, hf, hb);

    const dim3 gNode(cdiv_i(NNODE, 64), HIDP / 32);
    const dim3 gEdge(cdiv_i(ECH, 64), HIDP / 32);

    for (int i = 0; i < NI; ++i) {
        const __bf16* wm1i  = wm1b  + (size_t)i * NGP * HIDP;
        const __bf16* wm2i  = wm2b  + (size_t)i * HIDP * HIDP;
        const __bf16* wl1i  = wl1b  + (size_t)i * HIDP * HIDP;
        const __bf16* wl2i  = wl2b  + (size_t)i * HIDP * HIDP;
        const __bf16* wlini = wlinb + (size_t)i * HIDP * HIDP;
        const float* bm1i = bm1 + (size_t)i * HID;
        const float* bm2i = bm2 + (size_t)i * HID;
        const float* bl2i = bl2 + (size_t)i * HID;
        const float* blni = blin + (size_t)i * HID;

        // x1 = h @ Wl1 (no bias)
        k_gemm<0><<<gNode, T, 0, stream>>>(hb, wl1i, nullptr, nullptr, x1, nullptr,
                                           NNODE, HIDP, HIDP, HIDP, HIDP, HIDP);
        k_zero<<<cdiv_i((long)NNODE * HIDP, T), T, 0, stream>>>(agg, (long)NNODE * HIDP);

        for (int ch = 0; ch < NCH; ++ch) {
            const __bf16* Ae = eab + (size_t)ch * ECH * NGP;
            // t1 = ssp(edge_attr @ Wm1 + bm1)
            k_gemm<1><<<gEdge, T, 0, stream>>>(Ae, wm1i, bm1i, nullptr, nullptr, t1b,
                                               ECH, NGP, NGP, HIDP, HIDP, HIDP);
            // W = t1 @ Wm2 + bm2; agg[dst] += x1[src] * W * C  (fused scatter)
            k_gemm_scatter<<<gEdge, T, 0, stream>>>(t1b, wm2i, bm2i, Cg, x1,
                                                    srcp, dstp, agg, ECH, ch * ECH);
        }

        k_f2b<<<cdiv_i((long)NNODE * HIDP, T), T, 0, stream>>>(agg, aggb, (long)NNODE * HIDP);
        // t = ssp(agg @ Wl2 + bl2)
        k_gemm<1><<<gNode, T, 0, stream>>>(aggb, wl2i, bl2i, nullptr, nullptr, tb,
                                           NNODE, HIDP, HIDP, HIDP, HIDP, HIDP);
        // h = h + t @ Wlin + blin  (writes f32 h and bf16 h)
        k_gemm<2><<<gNode, T, 0, stream>>>(tb, wlini, blni, hf, hf, hb,
                                           NNODE, HIDP, HIDP, HIDP, HIDP, HIDP);
    }

    // per-graph mean pooling
    k_zero<<<cdiv_i((long)NBATCH * HIDP, T), T, 0, stream>>>(psum, (long)NBATCH * HIDP);
    k_zero<<<cdiv_i(NBATCH, T), T, 0, stream>>>(cnt, NBATCH);
    k_pool<<<cdiv_i((long)NNODE * HIDP, T), T, 0, stream>>>(hf, batch, psum, cnt);
    k_pdiv<<<cdiv_i((long)NBATCH * HIDP, T), T, 0, stream>>>(psum, cnt, pbf);

    // out = pooled @ Wpool + bpool  -> [128, 600] f32, tight ldo=600
    dim3 gOut(cdiv_i(NBATCH, 64), HIDP / 32);
    k_gemm<0><<<gOut, T, 0, stream>>>(pbf, wpb, bpool, nullptr, outp, nullptr,
                                      NBATCH, HIDP, HIDP, HIDP, HID, HID);
}
